// EmbedTrackLoss_37297495998726
// MI455X (gfx1250) — matrix operand order repaired and checked by reading, not compiled, not run on hardware
//
#include <hip/hip_runtime.h>

#define HH 512
#define WW 512
#define PP (HH * WW)
#define NI 8
#define NITEM 12
#define NB 2048
#define HTOT ((size_t)NITEM * NI * NB)

// pass1 launch geometry (fixed in kernel_launch): 64 blocks x 8 waves = 512 waves
#define P1_BLOCKS 64
#define P1_WAVES  (P1_BLOCKS * 8)        // 512
#define P1_ITERS  (PP / 32 / P1_WAVES)   // 8192 chunks / 512 waves = 16

typedef __attribute__((ext_vector_type(16))) _Float16 v16h;
typedef __attribute__((ext_vector_type(8)))  float    v8f;

__device__ __forceinline__ float frcp(float x) { return __builtin_amdgcn_rcpf(x); }
__device__ __forceinline__ float fsigmoid(float x) {
  return frcp(1.0f + __expf(-x));                       // v_exp + v_rcp, no div fixup
}
__device__ __forceinline__ float ftanh(float x) {
  const float xc = fminf(fmaxf(x, -15.0f), 15.0f);      // avoid inf*0
  const float e = __expf(2.0f * xc);
  return (e - 1.0f) * frcp(e + 1.0f);
}

// ---------------------------------------------------------------------------
// Kernel 0: zero workspace (harness poisons once; we must re-init every call)
// ---------------------------------------------------------------------------
__global__ void zero_ws_kernel(float* __restrict__ ws, int n) {
  for (int i = blockIdx.x * blockDim.x + threadIdx.x; i < n; i += gridDim.x * blockDim.x)
    ws[i] = 0.0f;
}

// ---------------------------------------------------------------------------
// Kernel 1: per-instance masked reductions as one-hot WMMA matmul.
//   D[inst][stat] += Onehot(16 x 32 pixels) * Vals(32 pixels x 16 stats)
//   stats: q0=npix q1=Ssig0 q2=Ssig1 q3=ccnt q4=Sy*c q5=Sx*c q6=Soffy*c q7=Soffx*c
// Also reduces seg seed_loss = sum_{label==0} seed^2.
// ---------------------------------------------------------------------------
__global__ __launch_bounds__(256) void pass1_stats_kernel(
    const float* __restrict__ segp, const float* __restrict__ trap,
    const int* __restrict__ inst, const unsigned char* __restrict__ cen,
    const int* __restrict__ labels, const float* __restrict__ offs,
    float* __restrict__ stats, float* __restrict__ seedloss) {
  const int item  = blockIdx.y;
  const bool isSeg = (item < 8);
  const int b     = isSeg ? item : (item - 8);
  const float* pred = isSeg ? (segp + (size_t)item * 5 * PP)
                            : (trap + (size_t)(item - 8) * 4 * PP);
  const int wave = threadIdx.x >> 5;
  const int lane = threadIdx.x & 31;

  __shared__ _Float16 s_val[8][32][8];
  __shared__ int      s_iid[8][32];
  __shared__ float    s_red[256];

  v8f acc = {0.f, 0.f, 0.f, 0.f, 0.f, 0.f, 0.f, 0.f};
  float seedacc = 0.0f;

  const int waveId = blockIdx.x * 8 + wave;    // 0..511

#pragma unroll 2
  for (int it = 0; it < P1_ITERS; ++it) {      // compile-time trip count -> 2 WMMAs/body
    const int c    = waveId + it * P1_WAVES;
    const int p    = c * 32 + lane;
    const int iid  = inst[(size_t)b * PP + p];
    const float cf = cen[(size_t)b * PP + p] ? 1.0f : 0.0f;
    const int y = p >> 9, x = p & (WW - 1);
    const float ym = y * (1.0f / (HH - 1));
    const float xm = x * (1.0f / (WW - 1));
    const float sg0 = fsigmoid(pred[(size_t)2 * PP + p]);
    const float sg1 = fsigmoid(pred[(size_t)3 * PP + p]);
    float offy = 0.0f, offx = 0.0f;
    if (!isSeg) {
      offy = offs[((size_t)b * 2 + 0) * PP + p];
      offx = offs[((size_t)b * 2 + 1) * PP + p];
    } else {
      const float seed = fsigmoid(pred[(size_t)4 * PP + p]);
      if (labels[(size_t)b * PP + p] == 0) seedacc += seed * seed;
    }

    __syncthreads();  // protect previous iteration's fragment reads
    s_iid[wave][lane] = iid;
    {
      _Float16* v = s_val[wave][lane];
      v[0] = (_Float16)1.0f;
      v[1] = (_Float16)sg0;
      v[2] = (_Float16)sg1;
      v[3] = (_Float16)cf;
      v[4] = (_Float16)(ym * cf);
      v[5] = (_Float16)(xm * cf);
      v[6] = (_Float16)(offy * cf);
      v[7] = (_Float16)(offx * cf);
    }
    __syncthreads();

    // A fragment (16x32 f16): lanes 0-15 row M=lane hold K {0..7,16..23};
    // lanes 16-31 row M=lane-16 hold K {8..15,24..31}.
    const int M  = lane & 15;
    const int kb = (lane < 16) ? 0 : 8;
    v16h afrag, bfrag;
#pragma unroll
    for (int j = 0; j < 8; ++j) {
      afrag[j]     = (s_iid[wave][kb + j]      == M + 1) ? (_Float16)1.0f : (_Float16)0.0f;
      afrag[j + 8] = (s_iid[wave][16 + kb + j] == M + 1) ? (_Float16)1.0f : (_Float16)0.0f;
    }
    // B fragment (32x16 f16): lane holds column N=lane%16; K offset by half-wave.
    const int q    = lane & 15;
    const int koff = (lane < 16) ? 0 : 8;
#pragma unroll
    for (int e = 0; e < 16; ++e)
      bfrag[e] = (q < 8) ? s_val[wave][(e < 8) ? (koff + e) : (16 + koff + e - 8)][q] : (_Float16)0.0f;

    acc = __builtin_amdgcn_wmma_f32_16x16x32_f16(false, afrag, false, bfrag,
                                                 (short)0, acc, false, false);
  }

  // Flush D tile: VGPR r -> row M = r + (lane<16 ? 0 : 8), col q = lane%16.
  {
    const int q  = lane & 15;
    const int mb = (lane < 16) ? 0 : 8;
#pragma unroll
    for (int r = 0; r < 8; ++r)
      atomicAdd(&stats[(size_t)item * 256 + (size_t)(mb + r) * 16 + q], acc[r]);
  }

  // block-reduce seed_loss (seg only)
  s_red[threadIdx.x] = seedacc;
  __syncthreads();
  for (int s = 128; s > 0; s >>= 1) {
    if (threadIdx.x < s) s_red[threadIdx.x] += s_red[threadIdx.x + s];
    __syncthreads();
  }
  if (threadIdx.x == 0 && isSeg) atomicAdd(&seedloss[item], s_red[0]);
}

// ---------------------------------------------------------------------------
// Kernel 2: derive per-(item,inst) quantities from WMMA stats.
// derived[t*16]: 0=cy 1=cx 2=s0 3=s1 4=smean0 5=smean1 6=valid 7=npix 8=gts
// ---------------------------------------------------------------------------
__global__ void derive_kernel(const float* __restrict__ stats, float* __restrict__ derived) {
  const int t = threadIdx.x;
  if (t >= NITEM * NI) return;
  const int item = t / NI, i = t % NI;
  const float* S = stats + (size_t)item * 256 + (size_t)i * 16;
  const float gts  = S[0];
  const float npix = fmaxf(gts, 1.0f);
  const float sm0 = S[1] / npix, sm1 = S[2] / npix;
  const float ccnt = S[3];
  float cy = S[4], cx = S[5];
  float valid = (fabsf(ccnt - 1.0f) < 0.25f) ? 1.0f : 0.0f;  // f16-exact small ints
  if (item >= 8) {
    cy -= S[6] * (1.0f / 256.0f);
    cx -= S[7] * (1.0f / 256.0f);
    if (fminf(cy, cx) < 0.0f || fmaxf(cy, cx) > 1.0f) valid = 0.0f;
  }
  float* D = derived + (size_t)t * 16;
  D[0] = cy; D[1] = cx;
  D[2] = __expf(10.0f * sm0);
  D[3] = __expf(10.0f * sm1);
  D[4] = sm0; D[5] = sm1;
  D[6] = valid; D[7] = npix; D[8] = gts;
}

// ---------------------------------------------------------------------------
// Kernel 3: heavy streaming pass: dist per (pixel, valid inst), error
// histograms (fg/bg counts + error sums), var_l and seed_l accumulators.
// Per-instance constants hoisted into registers (loop-invariant).
// ---------------------------------------------------------------------------
__global__ __launch_bounds__(256) void pass2_kernel(
    const float* __restrict__ segp, const float* __restrict__ trap,
    const int* __restrict__ inst, const float* __restrict__ derived,
    float* __restrict__ hc1, float* __restrict__ hc0,
    float* __restrict__ hs1, float* __restrict__ hs0,
    float* __restrict__ varl, float* __restrict__ seedl) {
  const int item  = blockIdx.y;
  const bool isSeg = (item < 8);
  const int b     = isSeg ? item : (item - 8);
  const float* pred = isSeg ? (segp + (size_t)item * 5 * PP)
                            : (trap + (size_t)(item - 8) * 4 * PP);
  __shared__ float lvar[NI], lseed[NI];
  if (threadIdx.x < NI) { lvar[threadIdx.x] = 0.0f; lseed[threadIdx.x] = 0.0f; }
  __syncthreads();

  // hoist per-instance constants into registers
  float rc0[NI], rc1[NI], rs0[NI], rs1[NI], rm0[NI], rm1[NI], rv[NI];
#pragma unroll
  for (int j = 0; j < NI; ++j) {
    const float* D = derived + ((size_t)item * NI + j) * 16;
    rc0[j] = D[0]; rc1[j] = D[1];
    rs0[j] = D[2]; rs1[j] = D[3];
    rm0[j] = D[4]; rm1[j] = D[5];
    rv[j]  = D[6];
  }

  const int stride = gridDim.x * blockDim.x;
  for (int p = blockIdx.x * blockDim.x + threadIdx.x; p < PP; p += stride) {
    __builtin_prefetch(&pred[(size_t)0 * PP + p + stride], 0, 1);  // global_prefetch_b8
    const int iid = inst[(size_t)b * PP + p];
    const int y = p >> 9, x = p & (WW - 1);
    const float ym = y * (1.0f / (HH - 1));
    const float xm = x * (1.0f / (WW - 1));
    const float t0 = ftanh(pred[(size_t)0 * PP + p]);
    const float t1 = ftanh(pred[(size_t)1 * PP + p]);
    const float e0 = isSeg ? (t0 + ym) : (ym - t0);
    const float e1 = isSeg ? (t1 + xm) : (xm - t1);
    const float sg0 = fsigmoid(pred[(size_t)2 * PP + p]);
    const float sg1 = fsigmoid(pred[(size_t)3 * PP + p]);
    const float seed = isSeg ? fsigmoid(pred[(size_t)4 * PP + p]) : 0.0f;

#pragma unroll
    for (int j = 0; j < NI; ++j) {
      if (rv[j] == 0.0f) continue;  // invalid instance -> contribution zeroed anyway
      const float dy = e0 - rc0[j];
      const float dx = e1 - rc1[j];
      const float dist = __expf(-(dy * dy * rs0[j] + dx * dx * rs1[j]));
      const bool fg = (iid == j + 1);
      const float err = fg ? (2.0f - 2.0f * dist) : (2.0f * dist);
      int bin = (int)(err * (NB * 0.5f));
      bin = bin < 0 ? 0 : (bin > NB - 1 ? NB - 1 : bin);
      const size_t hidx = ((size_t)item * NI + j) * NB + (size_t)bin;
      if (fg) {
        atomicAdd(&hc1[hidx], 1.0f);
        atomicAdd(&hs1[hidx], err);
        const float v0 = sg0 - rm0[j];
        const float v1 = sg1 - rm1[j];
        atomicAdd(&lvar[j], v0 * v0 + v1 * v1);
        if (isSeg) {
          const float sd = seed - dist;
          atomicAdd(&lseed[j], sd * sd);
        }
      } else {
        atomicAdd(&hc0[hidx], 1.0f);
        atomicAdd(&hs0[hidx], err);
      }
    }
  }
  __syncthreads();
  if (threadIdx.x < NI) {
    atomicAdd(&varl[(size_t)item * NI + threadIdx.x], lvar[threadIdx.x]);
    if (isSeg) atomicAdd(&seedl[(size_t)item * NI + threadIdx.x], lseed[threadIdx.x]);
  }
}

// ---------------------------------------------------------------------------
// Kernel 4: Lovász hinge via descending histogram scan (telescoped jac dot).
// ---------------------------------------------------------------------------
__global__ void lovasz_kernel(const float* __restrict__ hc1, const float* __restrict__ hc0,
                              const float* __restrict__ hs1, const float* __restrict__ hs0,
                              const float* __restrict__ derived, float* __restrict__ il) {
  const int t = threadIdx.x;
  if (t >= NITEM * NI) return;
  const float gts = derived[(size_t)t * 16 + 8];
  const size_t base = (size_t)t * NB;
  float cum1 = 0.0f, cum0 = 0.0f, jprev = 0.0f, loss = 0.0f;
  if (gts > 0.0f) {
    for (int bb = NB - 1; bb >= 0; --bb) {
      const float c1 = hc1[base + bb];
      const float c0 = hc0[base + bb];
      const float n = c1 + c0;
      if (n > 0.0f) {
        const float em = (hs1[base + bb] + hs0[base + bb]) / n;  // mean error in bin
        cum1 += c1; cum0 += c0;
        const float jac = 1.0f - (gts - cum1) / (gts + cum0);
        loss += em * (jac - jprev);  // errors >= 0 so relu is identity
        jprev = jac;
      }
    }
  }
  il[t] = loss * derived[(size_t)t * 16 + 6];  // * valid
}

// ---------------------------------------------------------------------------
// Kernel 5: combine everything into the scalar loss.
// ---------------------------------------------------------------------------
__global__ void finalize_kernel(const float* __restrict__ il, const float* __restrict__ varl,
                                const float* __restrict__ seedl, const float* __restrict__ seedloss,
                                const float* __restrict__ derived, float* __restrict__ out) {
  if (threadIdx.x != 0 || blockIdx.x != 0) return;
  float total = 0.0f;
  for (int item = 0; item < NITEM; ++item) {
    if (item < 8) {  // segmentation item
      float ils = 0.0f, vls = 0.0f, sls = 0.0f;
      for (int i = 0; i < NI; ++i) {
        const int t = item * NI + i;
        const float v = derived[(size_t)t * 16 + 6];
        const float npix = derived[(size_t)t * 16 + 7];
        ils += il[t];
        vls += (varl[t] / (2.0f * npix)) * v;  // / (N_SIGMA * npix)
        sls += seedl[t] * v;
      }
      const float seed_total = (seedloss[item] + 1.0f * sls) / (float)PP;
      total += 1.0f * ils + 10.0f * vls + 1.0f * seed_total;
    } else {  // tracking item
      float tls = 0.0f, vls = 0.0f, cnt = 0.0f;
      for (int i = 0; i < NI; ++i) {
        const int t = item * NI + i;
        const float v = derived[(size_t)t * 16 + 6];
        tls += il[t];
        vls += varl[t] * v;  // raw sum for tra
        cnt += v;
      }
      cnt = fmaxf(cnt, 1.0f);
      total += (1.0f * tls + 10.0f * vls) / cnt;
    }
  }
  out[0] = total;
}

// ---------------------------------------------------------------------------
extern "C" void kernel_launch(void* const* d_in, const int* in_sizes, int n_in,
                              void* d_out, int out_size, void* d_ws, size_t ws_size,
                              hipStream_t stream) {
  (void)in_sizes; (void)n_in; (void)out_size; (void)ws_size;
  const float*         segp   = (const float*)d_in[0];         // [8][5][512][512]
  const float*         trap   = (const float*)d_in[1];         // [4][4][512][512]
  const int*           inst   = (const int*)d_in[2];           // [8][512][512]
  const int*           labels = (const int*)d_in[3];           // [8][512][512]
  const unsigned char* cen    = (const unsigned char*)d_in[4]; // bool [8][512][512]
  const float*         offs   = (const float*)d_in[5];         // [4][2][512][512]
  float* out = (float*)d_out;
  float* ws  = (float*)d_ws;

  float* stats    = ws;                // 12*256     = 3072
  float* seedloss = stats + 3072;      // 12
  float* derived  = seedloss + 12;     // 12*8*16    = 1536
  float* varl     = derived + 1536;    // 96
  float* seedl    = varl + 96;         // 96
  float* il       = seedl + 96;        // 96
  float* hc1      = il + 96;           // 12*8*2048  = 196608
  float* hc0      = hc1 + HTOT;
  float* hs1      = hc0 + HTOT;
  float* hs0      = hs1 + HTOT;

  const int ztotal = 3072 + 12 + 1536 + 96 + 96 + 96 + 4 * (int)HTOT;  // ~791K floats
  zero_ws_kernel<<<512, 256, 0, stream>>>(ws, ztotal);

  dim3 g1(P1_BLOCKS, NITEM);
  pass1_stats_kernel<<<g1, 256, 0, stream>>>(segp, trap, inst, cen, labels, offs,
                                             stats, seedloss);
  derive_kernel<<<1, 128, 0, stream>>>(stats, derived);

  dim3 g2(256, NITEM);
  pass2_kernel<<<g2, 256, 0, stream>>>(segp, trap, inst, derived,
                                       hc1, hc0, hs1, hs0, varl, seedl);

  lovasz_kernel<<<1, 128, 0, stream>>>(hc1, hc0, hs1, hs0, derived, il);
  finalize_kernel<<<1, 64, 0, stream>>>(il, varl, seedl, seedloss, derived, out);
}
